// SpatialGraphConv_12695923326978
// MI455X (gfx1250) — matrix operand order repaired and checked by reading, not compiled, not run on hardware
//
#include <hip/hip_runtime.h>

typedef __attribute__((ext_vector_type(16))) __bf16 v16bf;
typedef __attribute__((ext_vector_type(8)))  float  v8f;

#define N_NODES 50000
#define N_EDGES 800000

// workspace layout (bytes)
#define XH_OFF   0                    // bf16 [N,128]  = 12,800,000 B
#define XT_OFF   12800000             // f32  [N,128]  = 25,600,000 B
#define W1P_OFF  38400000             // bf16 [8][8][32][16] = 65,536 B
#define WP_OFF   38465536             // bf16 [4][8][32][16] = 32,768 B

#if __has_builtin(__builtin_amdgcn_global_load_async_to_lds_b128) && \
    __has_builtin(__builtin_amdgcn_s_wait_asynccnt)
#define USE_ASYNC_LDS 1
#endif

#ifdef USE_ASYNC_LDS
typedef int v4i __attribute__((vector_size(16)));
typedef __attribute__((address_space(1))) v4i* g_v4i_p;   // global (AS1)
typedef __attribute__((address_space(3))) v4i* l_v4i_p;   // LDS (AS3)
#endif

union U16 { v16bf v; uint4 q[2]; };

__device__ __forceinline__ unsigned short f2bf(float f) {
    unsigned int u = __float_as_uint(f);
    u += 0x7fffu + ((u >> 16) & 1u);   // round-to-nearest-even
    return (unsigned short)(u >> 16);
}

__device__ __forceinline__ void atomic_fadd(float* p, float v) {
    __hip_atomic_fetch_add(p, v, __ATOMIC_RELAXED, __HIP_MEMORY_SCOPE_AGENT);
}

// Convert x -> bf16 copy, and zero the output buffer (same element count).
__global__ void prep_kernel(const float* __restrict__ x,
                            unsigned short* __restrict__ xh,
                            float* __restrict__ out, int n) {
    int i = blockIdx.x * blockDim.x + threadIdx.x;
    if (i < n) {
        xh[i]  = f2bf(x[i]);
        out[i] = 0.0f;
    }
}

// Repack W [128,128] and ew_W1 [256,128] (row-major [K,N]) into WMMA B-matrix
// lane layout: entry ((kk*8+nt)*32+lane)*16+e  holds  W[kk*32 + 8*(lane>=16) + e + 8*(e>=8)][nt*16 + lane%16]
__global__ void pack_kernel(const float* __restrict__ W,
                            const float* __restrict__ W1,
                            unsigned short* __restrict__ Wp,
                            unsigned short* __restrict__ W1p) {
    int i = blockIdx.x * blockDim.x + threadIdx.x;
    if (i < 32768) {                       // W1p: 8 K-steps
        int e = i & 15, lane = (i >> 4) & 31, nt = (i >> 9) & 7, kk = i >> 12;
        int col = nt * 16 + (lane & 15);
        int k   = kk * 32 + ((lane >> 4) ? 8 : 0) + e + ((e >> 3) ? 8 : 0);
        W1p[i] = f2bf(W1[k * 128 + col]);
    } else if (i < 49152) {                // Wp: 4 K-steps
        int j = i - 32768;
        int e = j & 15, lane = (j >> 4) & 31, nt = (j >> 9) & 7, kk = j >> 12;
        int col = nt * 16 + (lane & 15);
        int k   = kk * 32 + ((lane >> 4) ? 8 : 0) + e + ((e >> 3) ? 8 : 0);
        Wp[j] = f2bf(W[k * 128 + col]);
    }
}

// x_t = x @ W + b  (bf16 WMMA, fp32 accumulate). One wave per 16-row tile.
__global__ void xt_kernel(const unsigned short* __restrict__ xh,
                          const unsigned short* __restrict__ Wp,
                          const float* __restrict__ b,
                          float* __restrict__ xt) {
    int lane = threadIdx.x & 31;
    int wave = threadIdx.x >> 5;
    int tile = blockIdx.x * 8 + wave;
    if (tile >= N_NODES / 16) return;      // uniform per-wave exit
    int r  = lane & 15;
    int hi = lane >> 4;
    int lo = hi ? 8 : 0;

    const unsigned short* xrow = xh + (tile * 16 + r) * 128;
    U16 a[4];
#pragma unroll
    for (int kk = 0; kk < 4; kk++) {
        int koff = kk * 32 + lo;
        a[kk].q[0] = *(const uint4*)(xrow + koff);
        a[kk].q[1] = *(const uint4*)(xrow + koff + 16);
    }
#pragma unroll
    for (int nt = 0; nt < 8; nt++) {
        v8f acc = {0.f, 0.f, 0.f, 0.f, 0.f, 0.f, 0.f, 0.f};
#pragma unroll
        for (int kk = 0; kk < 4; kk++) {
            U16 bb;
            const unsigned short* bp = Wp + ((kk * 8 + nt) * 32 + lane) * 16;
            bb.q[0] = *(const uint4*)bp;
            bb.q[1] = *(const uint4*)(bp + 8);
            acc = __builtin_amdgcn_wmma_f32_16x16x32_bf16(
                false, a[kk].v, false, bb.v, (short)0, acc, false, false);
        }
        float bias = b[nt * 16 + r];
        // D layout: lane r/hi -> col nt*16+r, VGPR v -> row tile*16 + v + 8*hi
        float* op = xt + (tile * 16 + hi * 8) * 128 + nt * 16 + r;
#pragma unroll
        for (int v = 0; v < 8; v++) op[v * 128] = acc[v] + bias;
    }
}

// One wave per 16-edge tile: gather -> WMMA MLP (B from LDS) -> sigmoid -> atomic scatter.
__global__ void edge_kernel(const int* __restrict__ ei,   // [2,E]
                            const unsigned short* __restrict__ xh,
                            const unsigned short* __restrict__ W1p,
                            const float* __restrict__ b1,
                            const float* __restrict__ W2,
                            const float* __restrict__ b2,
                            const float* __restrict__ xt,
                            float* __restrict__ out) {
    // Stage the 64KB packed ew_W1 into LDS once per block (shared by 8 waves).
    __shared__ alignas(16) unsigned short w1s[32768];   // 64 KB
    {
        const uint4* g = (const uint4*)W1p;
        uint4* l = (uint4*)w1s;
#ifdef USE_ASYNC_LDS
        for (int i = threadIdx.x; i < 4096; i += 256) {
            __builtin_amdgcn_global_load_async_to_lds_b128(
                (g_v4i_p)(uintptr_t)(g + i),
                (l_v4i_p)(unsigned)(uintptr_t)(l + i), 0, 0);
        }
        __builtin_amdgcn_s_wait_asynccnt(0);
#else
        for (int i = threadIdx.x; i < 4096; i += 256) l[i] = g[i];
#endif
        __syncthreads();
    }

    int lane = threadIdx.x & 31;
    int wave = threadIdx.x >> 5;
    int tile = blockIdx.x * 8 + wave;
    if (tile >= N_EDGES / 16) return;      // uniform per-wave exit
    int r  = lane & 15;
    int hi = lane >> 4;
    int lo = hi ? 8 : 0;
    int e0 = tile * 16;

    int s_r = ei[e0 + r];                  // src of edge (e0 + r)
    int d_r = ei[N_EDGES + e0 + r];        // dst of edge (e0 + r)

    const unsigned short* srow = xh + s_r * 128;
    const unsigned short* drow = xh + d_r * 128;

    // A fragments: edge_feat[16, 256] = concat(x[src], x[dst]); kk 0-3 src, 4-7 dst.
    U16 a[8];
#pragma unroll
    for (int kk = 0; kk < 8; kk++) {
        const unsigned short* base = (kk < 4) ? srow : drow;
        int koff = (kk & 3) * 32 + lo;
        a[kk].q[0] = *(const uint4*)(base + koff);
        a[kk].q[1] = *(const uint4*)(base + koff + 16);
    }

    float ps[8];
#pragma unroll
    for (int v = 0; v < 8; v++) ps[v] = 0.0f;

#pragma unroll
    for (int nt = 0; nt < 8; nt++) {
        v8f acc = {0.f, 0.f, 0.f, 0.f, 0.f, 0.f, 0.f, 0.f};
#pragma unroll
        for (int kk = 0; kk < 8; kk++) {
            U16 bb;
            const unsigned short* bp = w1s + ((kk * 8 + nt) * 32 + lane) * 16;
            bb.q[0] = *(const uint4*)bp;                // ds_load_b128
            bb.q[1] = *(const uint4*)(bp + 8);
            acc = __builtin_amdgcn_wmma_f32_16x16x32_bf16(
                false, a[kk].v, false, bb.v, (short)0, acc, false, false);
        }
        float bias = b1[nt * 16 + r];
        float w2   = W2[nt * 16 + r];      // col = nt*16 + r
#pragma unroll
        for (int v = 0; v < 8; v++) {
            float h = acc[v] + bias;
            h = h > 0.0f ? h : 0.0f;       // ReLU
            ps[v] += h * w2;               // partial of h @ ew_W2 (this lane's column)
        }
    }

    // Row-sum across the 16 lanes of each half (rows v + 8*hi), then sigmoid.
#pragma unroll
    for (int v = 0; v < 8; v++) {
        float s = ps[v];
        s += __shfl_xor(s, 1);
        s += __shfl_xor(s, 2);
        s += __shfl_xor(s, 4);
        s += __shfl_xor(s, 8);
        ps[v] = 1.0f / (1.0f + __expf(-(s + b2[0])));
    }

    // Scatter: out[dst[e]] += w_e * x_t[src[e]]; each lane owns 4 contiguous floats.
    int j = lane * 4;
#pragma unroll
    for (int e = 0; e < 16; e++) {
        float w = __shfl(ps[e & 7], (e >> 3) * 16);   // row e = v + 8*hi
        int s   = __shfl(s_r, e);                     // lane e holds edge e0+e
        int d   = __shfl(d_r, e);
        float4 xv = *(const float4*)(xt + s * 128 + j);
        float* orow = out + d * 128 + j;
        atomic_fadd(orow + 0, w * xv.x);
        atomic_fadd(orow + 1, w * xv.y);
        atomic_fadd(orow + 2, w * xv.z);
        atomic_fadd(orow + 3, w * xv.w);
    }
}

extern "C" void kernel_launch(void* const* d_in, const int* in_sizes, int n_in,
                              void* d_out, int out_size, void* d_ws, size_t ws_size,
                              hipStream_t stream) {
    const float* x   = (const float*)d_in[0];
    const int*   ei  = (const int*)d_in[1];
    const float* W   = (const float*)d_in[2];
    const float* b   = (const float*)d_in[3];
    const float* W1  = (const float*)d_in[4];
    const float* b1  = (const float*)d_in[5];
    const float* W2  = (const float*)d_in[6];
    const float* b2  = (const float*)d_in[7];
    float* out = (float*)d_out;

    char* ws = (char*)d_ws;
    unsigned short* xh  = (unsigned short*)(ws + XH_OFF);
    float*          xt  = (float*)(ws + XT_OFF);
    unsigned short* W1p = (unsigned short*)(ws + W1P_OFF);
    unsigned short* Wp  = (unsigned short*)(ws + WP_OFF);

    int n = N_NODES * 128;                       // 6,400,000
    prep_kernel<<<(n + 255) / 256, 256, 0, stream>>>(x, xh, out, n);
    pack_kernel<<<192, 256, 0, stream>>>(W, W1, Wp, W1p);
    xt_kernel<<<(N_NODES / 16 + 7) / 8, 256, 0, stream>>>(xh, Wp, b, xt);
    edge_kernel<<<(N_EDGES / 16) / 8, 256, 0, stream>>>(ei, xh, W1p, b1, W2, b2, xt, out);
}
